// GATHeadV2_28123445854480
// MI455X (gfx1250) — matrix operand order repaired and checked by reading, not compiled, not run on hardware
//
#include <hip/hip_runtime.h>
#include <hip/hip_bf16.h>
#include <math.h>

// Model constants (match reference)
#define HEADS   4
#define HID     256
#define PROJ    1024
#define NGRAPHS 64

typedef __attribute__((ext_vector_type(16))) _Float16 v16h;
typedef __attribute__((ext_vector_type(8)))  float    v8f;

union FragH {
  v16h  v;
  uint4 q[2];
};

// LDS B-tile layout: 64 columns x 32 K f16, column stride padded to 40 f16
// (80 B) so 16 concurrent column reads hit distinct banks. 16B aligned.
#define BS_STRIDE 40
#define BS_ELEMS  (64 * BS_STRIDE)

// ---------------------------------------------------------------------------
// small helpers
// ---------------------------------------------------------------------------
__device__ __forceinline__ float atomicMaxF(float* addr, float value) {
  // classic trick: int compare for >=0, reversed unsigned compare for <0.
  if (value >= 0.0f)
    return __int_as_float(atomicMax((int*)addr, __float_as_int(value)));
  else
    return __uint_as_float(atomicMin((unsigned int*)addr, __float_as_uint(value)));
}

__device__ __forceinline__ float block_reduce_sum(float v, float* red) {
  int t = threadIdx.x;
  red[t] = v;
  __syncthreads();
  for (int s = blockDim.x >> 1; s > 0; s >>= 1) {
    if (t < s) red[t] += red[t + s];
    __syncthreads();
  }
  float r = red[0];
  __syncthreads();
  return r;
}

// ---------------------------------------------------------------------------
// conversions / init
// ---------------------------------------------------------------------------
__global__ void k_f32_to_f16(const float* __restrict__ src, _Float16* __restrict__ dst, long n) {
  long i = (long)blockIdx.x * blockDim.x + threadIdx.x;
  if (i < n) dst[i] = (_Float16)src[i];
}

// W: K x Nc row-major fp32  ->  Wt: Nc x K f16 (column-major for WMMA B frags)
__global__ void k_transpose_f16(const float* __restrict__ W, _Float16* __restrict__ Wt,
                                int K, int Nc) {
  long i = (long)blockIdx.x * blockDim.x + threadIdx.x;
  long total = (long)K * Nc;
  if (i >= total) return;
  int k = (int)(i / Nc);
  int n = (int)(i % Nc);
  Wt[(size_t)n * K + k] = (_Float16)W[i];
}

__global__ void k_init(float* __restrict__ p, float v, long n) {
  long i = (long)blockIdx.x * blockDim.x + threadIdx.x;
  if (i < n) p[i] = v;
}

// ---------------------------------------------------------------------------
// WMMA GEMM: C[M,Nc] = A[M,K] (f16 row-major) @ Bt[Nc,K] (f16, B pre-transposed)
// Block = 4 waves, 64x64 output tile. B tile staged in LDS, double-buffered and
// shared by all 4 waves; each wave holds 4 f32 accumulators (16x64 strip) and
// reuses its A fragment across the 4 WMMAs per k-step. The k-loop stages the
// next buffer unconditionally and the last k-step is peeled (no in-loop
// predicate -> straight-line body). No early returns (block barriers in the
// loop); out-of-range rows/cols are clamped and masked at the store.
// ---------------------------------------------------------------------------
__global__ void k_wmma_gemm(const _Float16* __restrict__ A,
                            const _Float16* __restrict__ Bt,
                            const float* __restrict__ bias,
                            float* __restrict__ C,
                            int M, int Nc, int K) {
  __shared__ __align__(16) _Float16 Bs[2][BS_ELEMS];

  const int lane = threadIdx.x;              // 0..31
  const int m    = lane & 15;                // A row in tile; B/C column in tile
  const int half = lane >> 4;                // K sub-chunk / output row group
  const int tid  = threadIdx.y * 32 + lane;  // 0..127

  const int ntM   = (M + 15) >> 4;
  int tileM = blockIdx.y * blockDim.y + threadIdx.y;
  const bool mvalid = (tileM < ntM);
  if (tileM >= ntM) tileM = ntM - 1;         // clamp; store is masked

  int arow = tileM * 16 + m;
  if (arow >= M) arow = M - 1;
  const _Float16* Arow = A + (size_t)arow * K;

  // B staging: 128 threads fill 64 cols x 32 K. thread -> (col, k-half).
  const int scol = tid >> 1;                 // 0..63
  const int skh  = tid & 1;                  // 0..1 (16 K values each)
  int sbc = blockIdx.x * 64 + scol;
  if (sbc >= Nc) sbc = Nc - 1;               // clamp: masked at store
  const _Float16* Bsrc = Bt + (size_t)sbc * K + skh * 16;
  _Float16* Bdst = &Bs[0][0] + scol * BS_STRIDE + skh * 16;

  auto stage = [&](int k0, int buf) {
    const _Float16* s = Bsrc + k0;
    uint4 u0 = *(const uint4*)(s);
    uint4 u1 = *(const uint4*)(s + 8);
    _Float16* d = Bdst + buf * BS_ELEMS;
    *(uint4*)(d)     = u0;
    *(uint4*)(d + 8) = u1;
  };

  v8f acc0 = {}, acc1 = {}, acc2 = {}, acc3 = {};

  auto step = [&](int k0, int buf) {
    FragH a;
    // A 16x32 f16 layout: lane holds row m; K = half*16..+15 (contiguous)
    a.q[0] = *(const uint4*)(Arow + k0 + (half << 4));
    a.q[1] = *(const uint4*)(Arow + k0 + 8 + (half << 4));

    // B fragments from LDS: lane holds column j*16+m, K = half*16..+15
    const _Float16* bbase = &Bs[buf][0] + half * 16;
    const _Float16* p0 = bbase + (0 * 16 + m) * BS_STRIDE;
    const _Float16* p1 = bbase + (1 * 16 + m) * BS_STRIDE;
    const _Float16* p2 = bbase + (2 * 16 + m) * BS_STRIDE;
    const _Float16* p3 = bbase + (3 * 16 + m) * BS_STRIDE;
    FragH b0, b1, b2, b3;
    b0.q[0] = *(const uint4*)p0; b0.q[1] = *(const uint4*)(p0 + 8);
    b1.q[0] = *(const uint4*)p1; b1.q[1] = *(const uint4*)(p1 + 8);
    b2.q[0] = *(const uint4*)p2; b2.q[1] = *(const uint4*)(p2 + 8);
    b3.q[0] = *(const uint4*)p3; b3.q[1] = *(const uint4*)(p3 + 8);

    acc0 = __builtin_amdgcn_wmma_f32_16x16x32_f16(false, a.v, false, b0.v, (short)0, acc0, false, false);
    acc1 = __builtin_amdgcn_wmma_f32_16x16x32_f16(false, a.v, false, b1.v, (short)0, acc1, false, false);
    acc2 = __builtin_amdgcn_wmma_f32_16x16x32_f16(false, a.v, false, b2.v, (short)0, acc2, false, false);
    acc3 = __builtin_amdgcn_wmma_f32_16x16x32_f16(false, a.v, false, b3.v, (short)0, acc3, false, false);
  };

  stage(0, 0);
  __syncthreads();

  // main loop: stage next buffer unconditionally; last k-step peeled below.
  const int kLast = K - 32;                  // K is a multiple of 32 (512/1024)
  int buf = 0;
  for (int k0 = 0; k0 < kLast; k0 += 32) {
    stage(k0 + 32, buf ^ 1);                 // overlap next stage with WMMAs
    __builtin_prefetch(Arow + k0 + 64 + (half << 3), 0, 1);
    step(k0, buf);
    __syncthreads();                         // buf^1 staged; buf reads done
    buf ^= 1;
  }
  step(kLast, buf);                          // tail: consume only

  // D layout: col = lane&15, row = tileM*16 + half*8 + r for VGPR r
  if (mvalid) {
    const v8f accs[4] = {acc0, acc1, acc2, acc3};
    #pragma unroll
    for (int j = 0; j < 4; ++j) {
      const int ccol = blockIdx.x * 64 + j * 16 + m;
      if (ccol < Nc) {
        const float bv = bias ? bias[ccol] : 0.0f;
        #pragma unroll
        for (int r = 0; r < 8; ++r) {
          const int crow = tileM * 16 + half * 8 + r;
          if (crow < M) C[(size_t)crow * Nc + ccol] = accs[j][r] + bv;
        }
      }
    }
  }
}

// ---------------------------------------------------------------------------
// alpha_s / alpha_d: per (node,head) dot of h[node,head,:] with att vectors.
// grid = N*HEADS waves of 32 threads.
// ---------------------------------------------------------------------------
__global__ void k_alpha(const float* __restrict__ H,
                        const float* __restrict__ att_s,
                        const float* __restrict__ att_d,
                        float* __restrict__ as_, float* __restrict__ ad_,
                        int nNodes) {
  const int nh   = blockIdx.x;
  const int node = nh >> 2;
  const int head = nh & 3;
  if (node >= nNodes) return;
  const float* hp = H + (size_t)node * PROJ + head * HID;
  float ps = 0.0f, pd = 0.0f;
  for (int j = threadIdx.x; j < HID; j += 32) {
    const float v = hp[j];
    ps += v * att_s[head * HID + j];
    pd += v * att_d[head * HID + j];
  }
  for (int off = 16; off > 0; off >>= 1) {
    ps += __shfl_down(ps, off, 32);
    pd += __shfl_down(pd, off, 32);
  }
  if (threadIdx.x == 0) {
    as_[node * HEADS + head] = ps;
    ad_[node * HEADS + head] = pd;
  }
}

// ---------------------------------------------------------------------------
// edge phase (ET = E real edges + N self-loops appended)
// ---------------------------------------------------------------------------
__global__ void k_edge_max(const int* __restrict__ ei,
                           const float* __restrict__ as_, const float* __restrict__ ad_,
                           float* __restrict__ esc, float* __restrict__ smax,
                           int E, int ET) {
  long i = (long)blockIdx.x * blockDim.x + threadIdx.x;
  if (i >= (long)ET * HEADS) return;
  const int e = (int)(i >> 2);
  const int h = (int)(i & 3);
  const int src = (e < E) ? ei[e]     : (e - E);
  const int dst = (e < E) ? ei[E + e] : (e - E);
  float s = as_[src * HEADS + h] + ad_[dst * HEADS + h];
  s = (s > 0.0f) ? s : 0.2f * s;                       // leaky_relu 0.2
  esc[(size_t)e * HEADS + h] = s;
  atomicMaxF(&smax[dst * HEADS + h], s);
}

__global__ void k_edge_exp(const int* __restrict__ ei,
                           float* __restrict__ esc,
                           const float* __restrict__ smax,
                           float* __restrict__ ssum,
                           int E, int ET) {
  long i = (long)blockIdx.x * blockDim.x + threadIdx.x;
  if (i >= (long)ET * HEADS) return;
  const int e = (int)(i >> 2);
  const int h = (int)(i & 3);
  const int dst = (e < E) ? ei[E + e] : (e - E);
  const float w = expf(esc[(size_t)e * HEADS + h] - smax[dst * HEADS + h]);
  esc[(size_t)e * HEADS + h] = w;
  atomicAdd(&ssum[dst * HEADS + h], w);
}

__global__ void k_scatter(const int* __restrict__ ei,
                          const float* __restrict__ H,
                          const float* __restrict__ esc,
                          const float* __restrict__ ssum,
                          float* __restrict__ out,
                          int E, int ET) {
  __shared__ float wsh[HEADS];
  const int e = blockIdx.x;
  if (e >= ET) return;
  const int src = (e < E) ? ei[e]     : (e - E);
  const int dst = (e < E) ? ei[E + e] : (e - E);
  if (threadIdx.x < HEADS)
    wsh[threadIdx.x] = esc[(size_t)e * HEADS + threadIdx.x] /
                       (ssum[dst * HEADS + threadIdx.x] + 1e-16f);
  __syncthreads();
  const float* hs = H + (size_t)src * PROJ;
  float*       od = out + (size_t)dst * PROJ;
  for (int d = threadIdx.x; d < PROJ; d += blockDim.x)
    atomicAdd(&od[d], hs[d] * wsh[d >> 8]);            // HID = 256
}

// ---------------------------------------------------------------------------
// fused:  x = LayerNorm(x + elu(agg + gat_bias)) * gamma + beta
// writes fp32 (for residual/pool) and f16 (for next GEMM). 256 thr / node.
// ---------------------------------------------------------------------------
__global__ void k_elu_ln(const float* __restrict__ agg,
                         const float* __restrict__ gbias,
                         const float* __restrict__ gamma,
                         const float* __restrict__ beta,
                         float* __restrict__ Xf, _Float16* __restrict__ Xh,
                         int nNodes) {
  __shared__ float red[256];
  const int node = blockIdx.x;
  if (node >= nNodes) return;
  const size_t base = (size_t)node * PROJ;
  float loc[4];
  float sum = 0.0f;
  #pragma unroll
  for (int i = 0; i < 4; ++i) {
    const int d = threadIdx.x + i * 256;
    float t = agg[base + d] + gbias[d];
    t = (t > 0.0f) ? t : expm1f(t);                    // elu
    const float v = Xf[base + d] + t;                  // residual
    loc[i] = v;
    sum += v;
  }
  const float mu = block_reduce_sum(sum, red) * (1.0f / PROJ);
  float ss = 0.0f;
  #pragma unroll
  for (int i = 0; i < 4; ++i) { const float dlt = loc[i] - mu; ss += dlt * dlt; }
  const float var  = block_reduce_sum(ss, red) * (1.0f / PROJ);
  const float rstd = rsqrtf(var + 1e-5f);
  #pragma unroll
  for (int i = 0; i < 4; ++i) {
    const int d = threadIdx.x + i * 256;
    const float o = (loc[i] - mu) * rstd * gamma[d] + beta[d];
    Xf[base + d] = o;
    Xh[base + d] = (_Float16)o;
  }
}

// ---------------------------------------------------------------------------
// pooling
// ---------------------------------------------------------------------------
// scores[n] = sum_j tanh(S1[n,j]) * W2[j] + b2   (S1 already has b1 from GEMM)
__global__ void k_pool_score(const float* __restrict__ S1,
                             const float* __restrict__ W2,
                             const float* __restrict__ b2,
                             float* __restrict__ scores, int nNodes) {
  __shared__ float red[256];
  const int node = blockIdx.x;
  if (node >= nNodes) return;
  float p = 0.0f;
  for (int d = threadIdx.x; d < PROJ / 2; d += blockDim.x)
    p += tanhf(S1[(size_t)node * (PROJ / 2) + d]) * W2[d];
  const float tot = block_reduce_sum(p, red);
  if (threadIdx.x == 0) scores[node] = tot + b2[0];
}

__global__ void k_graph_max(const int* __restrict__ batch,
                            const float* __restrict__ scores,
                            float* __restrict__ gmax, int nNodes) {
  int i = blockIdx.x * blockDim.x + threadIdx.x;
  if (i < nNodes) atomicMaxF(&gmax[batch[i]], scores[i]);
}

__global__ void k_graph_expsum(const int* __restrict__ batch,
                               const float* __restrict__ scores,
                               const float* __restrict__ gmax,
                               float* __restrict__ wnode,
                               float* __restrict__ gsum, int nNodes) {
  int i = blockIdx.x * blockDim.x + threadIdx.x;
  if (i >= nNodes) return;
  const int b = batch[i];
  const float w = expf(scores[i] - gmax[b]);
  wnode[i] = w;
  atomicAdd(&gsum[b], w);
}

__global__ void k_pool_scatter(const int* __restrict__ batch,
                               const float* __restrict__ Xf,
                               const float* __restrict__ wnode,
                               const float* __restrict__ gsum,
                               float* __restrict__ pooled, int nNodes) {
  const int node = blockIdx.x;
  if (node >= nNodes) return;
  const int b = batch[node];
  const float w = wnode[node] / (gsum[b] + 1e-16f);
  const float* xp = Xf + (size_t)node * PROJ;
  float*       pp = pooled + (size_t)b * PROJ;
  for (int d = threadIdx.x; d < PROJ; d += blockDim.x)
    atomicAdd(&pp[d], xp[d] * w);
}

// exact GELU, emit f16 for next GEMM
__global__ void k_gelu_to_f16(const float* __restrict__ in, _Float16* __restrict__ out, long n) {
  long i = (long)blockIdx.x * blockDim.x + threadIdx.x;
  if (i >= n) return;
  const float t = in[i];
  out[i] = (_Float16)(0.5f * t * (1.0f + erff(t * 0.70710678118654752f)));
}

// ---------------------------------------------------------------------------
// host launcher
// ---------------------------------------------------------------------------
extern "C" void kernel_launch(void* const* d_in, const int* in_sizes, int n_in,
                              void* d_out, int out_size, void* d_ws, size_t ws_size,
                              hipStream_t stream) {
  (void)n_in; (void)out_size; (void)ws_size;

  const float* x       = (const float*)d_in[0];
  const int*   ei      = (const int*)d_in[1];
  const int*   batch   = (const int*)d_in[2];
  const float* proj_W  = (const float*)d_in[3];
  const float* proj_b  = (const float*)d_in[4];
  const float* gat_W   = (const float*)d_in[5];
  const float* att_src = (const float*)d_in[6];
  const float* att_dst = (const float*)d_in[7];
  const float* gat_b   = (const float*)d_in[8];
  const float* gamma   = (const float*)d_in[9];
  const float* beta    = (const float*)d_in[10];
  const float* pool_W1 = (const float*)d_in[11];
  const float* pool_b1 = (const float*)d_in[12];
  const float* pool_W2 = (const float*)d_in[13];
  const float* pool_b2 = (const float*)d_in[14];
  const float* cls_W1  = (const float*)d_in[15];
  const float* cls_b1  = (const float*)d_in[16];
  const float* cls_W2  = (const float*)d_in[17];
  const float* cls_b2  = (const float*)d_in[18];
  float* out = (float*)d_out;

  const int N  = in_sizes[0] / PROJ;   // 10000
  const int E  = in_sizes[1] / 2;      // 160000
  const int ET = E + N;                // edges + self loops

  // ---- workspace carving (256B aligned) ----
  char* ws = (char*)d_ws;
  auto carve = [&](size_t bytes) {
    char* p = ws;
    ws += (bytes + 255) & ~(size_t)255;
    return (void*)p;
  };
  _Float16* Xh     = (_Float16*)carve((size_t)N * PROJ * 2);     // f16 activations
  _Float16* Wt     = (_Float16*)carve((size_t)PROJ * PROJ * 2);  // transposed weight (f16)
  float*    Xf     = (float*)carve((size_t)N * PROJ * 4);        // fp32 activations
  float*    Hf     = (float*)carve((size_t)N * PROJ * 4);        // GEMM out / S1 reuse
  float*    Af     = (float*)carve((size_t)N * PROJ * 4);        // aggregation target
  float*    aS     = (float*)carve((size_t)N * HEADS * 4);
  float*    aD     = (float*)carve((size_t)N * HEADS * 4);
  float*    esc    = (float*)carve((size_t)ET * HEADS * 4);
  float*    smax   = (float*)carve((size_t)N * HEADS * 4);
  float*    ssum   = (float*)carve((size_t)N * HEADS * 4);
  float*    scores = (float*)carve((size_t)N * 4);
  float*    wnode  = (float*)carve((size_t)N * 4);
  float*    gmax   = (float*)carve(NGRAPHS * 4);
  float*    gsum   = (float*)carve(NGRAPHS * 4);
  float*    pooled = (float*)carve((size_t)NGRAPHS * PROJ * 4);
  _Float16* pldH   = (_Float16*)carve((size_t)NGRAPHS * PROJ * 2);
  float*    hid    = (float*)carve((size_t)NGRAPHS * (PROJ / 2) * 4);
  _Float16* hidH   = (_Float16*)carve((size_t)NGRAPHS * (PROJ / 2) * 2);

  auto cvt = [&](const float* s, _Float16* d, long n) {
    k_f32_to_f16<<<dim3((unsigned)((n + 255) / 256)), dim3(256), 0, stream>>>(s, d, n);
  };
  auto trans = [&](const float* W, _Float16* D, int K, int Nc) {
    long n = (long)K * Nc;
    k_transpose_f16<<<dim3((unsigned)((n + 255) / 256)), dim3(256), 0, stream>>>(W, D, K, Nc);
  };
  auto gemm = [&](const _Float16* A, const _Float16* Bt, const float* bias,
                  float* C, int M, int Nc, int K) {
    const int ntN = (Nc + 15) / 16;          // 16-wide N tiles
    const int ntM = (M + 15) / 16;           // 16-wide M tiles
    dim3 grid((unsigned)((ntN + 3) / 4),     // 64-wide N per block
              (unsigned)((ntM + 3) / 4));    // 4 waves (M-tiles) per block
    k_wmma_gemm<<<grid, dim3(32, 4), 0, stream>>>(A, Bt, bias, C, M, Nc, K);
  };
  auto fill = [&](float* p, float v, long n) {
    k_init<<<dim3((unsigned)((n + 255) / 256)), dim3(256), 0, stream>>>(p, v, n);
  };

  // ---- input projection: Xf = x @ proj_W + proj_b ----
  cvt(x, Xh, (long)N * PROJ);
  trans(proj_W, Wt, PROJ, PROJ);
  gemm(Xh, Wt, proj_b, Xf, N, PROJ, PROJ);
  cvt(Xf, Xh, (long)N * PROJ);

  // ---- 3 GAT layers ----
  for (int L = 0; L < 3; ++L) {
    trans(gat_W + (size_t)L * PROJ * PROJ, Wt, PROJ, PROJ);
    gemm(Xh, Wt, nullptr, Hf, N, PROJ, PROJ);          // h = x @ W

    k_alpha<<<dim3((unsigned)(N * HEADS)), dim3(32), 0, stream>>>(
        Hf, att_src + (size_t)L * HEADS * HID, att_dst + (size_t)L * HEADS * HID,
        aS, aD, N);

    fill(smax, -INFINITY, (long)N * HEADS);
    fill(ssum, 0.0f, (long)N * HEADS);
    fill(Af, 0.0f, (long)N * PROJ);

    const long nEH = (long)ET * HEADS;
    k_edge_max<<<dim3((unsigned)((nEH + 255) / 256)), dim3(256), 0, stream>>>(
        ei, aS, aD, esc, smax, E, ET);
    k_edge_exp<<<dim3((unsigned)((nEH + 255) / 256)), dim3(256), 0, stream>>>(
        ei, esc, smax, ssum, E, ET);
    k_scatter<<<dim3((unsigned)ET), dim3(256), 0, stream>>>(
        ei, Hf, esc, ssum, Af, E, ET);

    k_elu_ln<<<dim3((unsigned)N), dim3(256), 0, stream>>>(
        Af, gat_b + (size_t)L * PROJ,
        gamma + (size_t)L * PROJ, beta + (size_t)L * PROJ,
        Xf, Xh, N);
  }

  // ---- attention pooling ----
  trans(pool_W1, Wt, PROJ, PROJ / 2);
  gemm(Xh, Wt, pool_b1, Hf, N, PROJ / 2, PROJ);        // S1 = x@W1 + b1 (reuse Hf)
  k_pool_score<<<dim3((unsigned)N), dim3(256), 0, stream>>>(Hf, pool_W2, pool_b2, scores, N);

  fill(gmax, -INFINITY, NGRAPHS);
  fill(gsum, 0.0f, NGRAPHS);
  fill(pooled, 0.0f, (long)NGRAPHS * PROJ);
  k_graph_max<<<dim3((unsigned)((N + 255) / 256)), dim3(256), 0, stream>>>(batch, scores, gmax, N);
  k_graph_expsum<<<dim3((unsigned)((N + 255) / 256)), dim3(256), 0, stream>>>(
      batch, scores, gmax, wnode, gsum, N);
  k_pool_scatter<<<dim3((unsigned)N), dim3(256), 0, stream>>>(batch, Xf, wnode, gsum, pooled, N);

  // ---- classifier head ----
  cvt(pooled, pldH, (long)NGRAPHS * PROJ);
  trans(cls_W1, Wt, PROJ, PROJ / 2);
  gemm(pldH, Wt, cls_b1, hid, NGRAPHS, PROJ / 2, PROJ);
  k_gelu_to_f16<<<dim3((unsigned)((NGRAPHS * (PROJ / 2) + 255) / 256)), dim3(256), 0, stream>>>(
      hid, hidH, (long)NGRAPHS * (PROJ / 2));
  trans(cls_W2, Wt, PROJ / 2, 14);
  gemm(hidH, Wt, cls_b2, out, NGRAPHS, 14, PROJ / 2);
}
